// FP4Linear_28467043238108
// MI455X (gfx1250) — compile-verified, matrix operand order checked by reading, not run on hardware
//
#include <hip/hip_runtime.h>

typedef __attribute__((ext_vector_type(16))) __bf16 v16bf;
typedef __attribute__((ext_vector_type(8)))  float  v8f;

#define IN_F  4096
#define OUT_F 11008
#define BM 128
#define BN 128
#define BK 64
#define NK (IN_F / BK)

// Pack two fp32 -> packed bf16 pair by truncation: one v_perm_b32.
// result[15:0] = a[31:16], result[31:16] = b[31:16]
__device__ __forceinline__ unsigned pkt(float a, float b) {
    return __builtin_amdgcn_perm(__builtin_bit_cast(unsigned, b),
                                 __builtin_bit_cast(unsigned, a), 0x07060302u);
}

union FragU { v16bf v; uint4 q[2]; };
union AccU  { v8f v; float f[8]; };

__global__ __launch_bounds__(256)
void fp4linear_wmma_kernel(const float* __restrict__ x,
                           const int* __restrict__ codes,
                           const float* __restrict__ absmax,
                           const float* __restrict__ bias,
                           float* __restrict__ out)
{
    __shared__ unsigned short lds_a[2][BM * BK];  // x tile, bf16, [m][k]
    __shared__ unsigned short lds_b[2][BN * BK];  // W tile, bf16, [n][k]
    __shared__ float lut[16];

    const int t = threadIdx.x;
    if (t == 0) {
        lut[0]  =  0.0f;          lut[1]  =  0.0052083333f;
        lut[2]  =  0.6666666666f; lut[3]  =  1.0f;
        lut[4]  =  0.3333333333f; lut[5]  =  0.5f;
        lut[6]  =  0.1666666666f; lut[7]  =  0.25f;
        lut[8]  = -0.0f;          lut[9]  = -0.0052083333f;
        lut[10] = -0.6666666666f; lut[11] = -1.0f;
        lut[12] = -0.3333333333f; lut[13] = -0.5f;
        lut[14] = -0.1666666666f; lut[15] = -0.25f;
    }

    const int nbase = blockIdx.x * BN;
    const int mbase = blockIdx.y * BM;

    const int lane = t & 31;
    const int wave = t >> 5;
    const int wm = (wave & 3) * 32;    // wave's M offset in tile (2 frags)
    const int wn = (wave >> 2) * 64;   // wave's N offset in tile (4 frags)

    // staging coords: 2 threads per row, each covers a 32-wide K strip
    const int row  = t >> 1;           // 0..127
    const int koff = (t & 1) * 32;     // stays inside one absmax-64 block

    // init accumulators with bias (bias depends only on N)
    AccU acc[2][4];
    #pragma unroll
    for (int ni = 0; ni < 4; ++ni) {
        float b = bias[nbase + wn + ni * 16 + (lane & 15)];
        #pragma unroll
        for (int mi = 0; mi < 2; ++mi)
            #pragma unroll
            for (int r = 0; r < 8; ++r) acc[mi][ni].f[r] = b;
    }

    // per-lane K chunk offsets for 16-bit A/B fragments (ISA 7.12.2)
    const int c0 = (lane >> 4) * 8;
    const int c1 = 16 + (lane >> 4) * 8;

    const float* asrc = x + (size_t)(mbase + row) * IN_F + koff;
    const int*   csrc = codes + (size_t)(nbase + row) * IN_F + koff;
    const float* amp  = absmax + (size_t)(nbase + row) * (IN_F / 64);

    auto stage = [&](int buf, int itk) {
        const float4* s  = (const float4*)(asrc + itk * BK);
        const int4*   cs = (const int4*)(csrc + itk * BK);
        float4 fa[8];
        int4   qa[8];
        #pragma unroll
        for (int j = 0; j < 8; ++j) fa[j] = s[j];
        #pragma unroll
        for (int j = 0; j < 8; ++j) qa[j] = cs[j];
        const float amax = amp[itk];   // itk*BK >> 6 == itk for BK=64

        uint4* da = (uint4*)&lds_a[buf][row * BK + koff];
        uint4* db = (uint4*)&lds_b[buf][row * BK + koff];
        #pragma unroll
        for (int j = 0; j < 4; ++j) {
            uint4 u;
            u.x = pkt(fa[2*j].x,   fa[2*j].y);
            u.y = pkt(fa[2*j].z,   fa[2*j].w);
            u.z = pkt(fa[2*j+1].x, fa[2*j+1].y);
            u.w = pkt(fa[2*j+1].z, fa[2*j+1].w);
            da[j] = u;
        }
        #pragma unroll
        for (int j = 0; j < 4; ++j) {
            const int4 a = qa[2*j], b = qa[2*j+1];
            uint4 u;
            u.x = pkt(lut[a.x & 15] * amax, lut[a.y & 15] * amax);
            u.y = pkt(lut[a.z & 15] * amax, lut[a.w & 15] * amax);
            u.z = pkt(lut[b.x & 15] * amax, lut[b.y & 15] * amax);
            u.w = pkt(lut[b.z & 15] * amax, lut[b.w & 15] * amax);
            db[j] = u;
        }
    };

    // ---------- pipelined main loop: 1 barrier, 16 WMMAs per K-step ----------
    stage(0, 0);

    for (int it = 0; it < NK; ++it) {
        __syncthreads();   // stage(it) visible; compute(it-1) done before reuse

        const int nxt = it + 1;
        if (nxt < NK) {
            if (nxt + 1 < NK) {
                __builtin_prefetch(asrc + (nxt + 1) * BK, 0, 0);
                __builtin_prefetch(csrc + (nxt + 1) * BK, 0, 0);
            }
            stage(nxt & 1, nxt);
        }

        const int cur = it & 1;
        #pragma unroll
        for (int kk = 0; kk < BK; kk += 32) {
            // Load ALL fragments for this K-half into distinct registers so
            // the 12 ds_load_b128s overlap and the 8 WMMAs issue back-to-back.
            FragU af[2], bg[4];
            #pragma unroll
            for (int mi = 0; mi < 2; ++mi) {
                const int m = wm + mi * 16 + (lane & 15);
                af[mi].q[0] = *(const uint4*)&lds_a[cur][m * BK + kk + c0];
                af[mi].q[1] = *(const uint4*)&lds_a[cur][m * BK + kk + c1];
            }
            #pragma unroll
            for (int ni = 0; ni < 4; ++ni) {
                const int n = wn + ni * 16 + (lane & 15);
                bg[ni].q[0] = *(const uint4*)&lds_b[cur][n * BK + kk + c0];
                bg[ni].q[1] = *(const uint4*)&lds_b[cur][n * BK + kk + c1];
            }
            #pragma unroll
            for (int ni = 0; ni < 4; ++ni)
                #pragma unroll
                for (int mi = 0; mi < 2; ++mi)
                    acc[mi][ni].v = __builtin_amdgcn_wmma_f32_16x16x32_bf16(
                        false, af[mi].v, false, bg[ni].v,
                        (short)0, acc[mi][ni].v, false, false);
        }
    }

    // ---- store D: VGPR r, lane L -> M = r + 8*(L/16), N = L%16 ----
    #pragma unroll
    for (int mi = 0; mi < 2; ++mi) {
        #pragma unroll
        for (int ni = 0; ni < 4; ++ni) {
            const int n  = nbase + wn + ni * 16 + (lane & 15);
            const int mb = mbase + wm + mi * 16 + (lane >> 4) * 8;
            #pragma unroll
            for (int r = 0; r < 8; ++r)
                out[(size_t)(mb + r) * OUT_F + n] = acc[mi][ni].f[r];
        }
    }
}

extern "C" void kernel_launch(void* const* d_in, const int* in_sizes, int n_in,
                              void* d_out, int out_size, void* d_ws, size_t ws_size,
                              hipStream_t stream) {
    (void)n_in; (void)out_size; (void)d_ws; (void)ws_size;
    const float* x      = (const float*)d_in[0];
    const int*   codes  = (const int*)d_in[1];
    const float* absmax = (const float*)d_in[2];
    const float* bias   = (const float*)d_in[3];
    float*       out    = (float*)d_out;

    const int M = in_sizes[0] / IN_F;          // 4*2048 = 8192
    dim3 grid(OUT_F / BN, M / BM);             // 86 x 64
    dim3 block(256);
    fp4linear_wmma_kernel<<<grid, block, 0, stream>>>(x, codes, absmax, bias, out);
}
